// attention_19224273617398
// MI455X (gfx1250) — matrix operand order repaired and checked by reading, not compiled, not run on hardware
//
#include <hip/hip_runtime.h>
#include <hip/hip_bf16.h>
#include <stdint.h>

typedef __attribute__((ext_vector_type(16))) __bf16   v16bf;
typedef __attribute__((ext_vector_type(2)))  __bf16   v2bf;
typedef __attribute__((ext_vector_type(8)))  float    v8f;
typedef __attribute__((ext_vector_type(8)))  unsigned v8u;
typedef __attribute__((ext_vector_type(4)))  unsigned u32x4;
typedef __attribute__((ext_vector_type(8)))  int      i32x8;
typedef __attribute__((ext_vector_type(4)))  int      i32x4;

#define WMMA_BF16(A, B, C) \
  __builtin_amdgcn_wmma_f32_16x16x32_bf16(false, (A), false, (B), (short)0, (C), false, false)

// 1/sqrt(64) * log2(e): folded into Q at projection time; softmax in base-2 domain.
#define SM_SCALE 0.18033688011112042f

__device__ __forceinline__ unsigned pack_bf16x2(float lo, float hi) {
  v2bf t;
  t[0] = (__bf16)lo;          // native RNE converts; pair should fuse to v_cvt_pk_bf16_f32
  t[1] = (__bf16)hi;
  return __builtin_bit_cast(unsigned, t);
}

__device__ __forceinline__ unsigned short f32_to_bf16(float f) {
  __bf16 t = (__bf16)f;
  return __builtin_bit_cast(unsigned short, t);
}

__device__ __forceinline__ v16bf pack16(uint4 x0, uint4 x1) {
  v8u r = {x0.x, x0.y, x0.z, x0.w, x1.x, x1.y, x1.z, x1.w};
  return __builtin_bit_cast(v16bf, r);
}

// A operand, matrix row-major [m][k] (ld ushorts). Lane L: row m=L%16;
// two contiguous 16B runs at k0+8hf and k0+16+8hf.
__device__ __forceinline__ v16bf load_matA(const unsigned short* p, int ld, int k0, int lane) {
  int m = lane & 15, hf = (lane >> 4) & 1;
  const unsigned short* row = p + m * ld + k0 + 8 * hf;
  uint4 x0 = *(const uint4*)(row);
  uint4 x1 = *(const uint4*)(row + 16);
  return pack16(x0, x1);
}

// B operand from matrix stored [n][k] row-major. Lane L: col n=L%16;
// two contiguous 16B runs at k0+8hf and k0+16+8hf.
__device__ __forceinline__ v16bf load_matBt(const unsigned short* p, int ld, int k0, int lane) {
  int n = lane & 15, hf = (lane >> 4) & 1;
  const uint4* q = (const uint4*)(p + n * ld + k0 + 8 * hf);
  return pack16(q[0], q[2]);
}

// ---- Tensor Data Mover: 2D tile (data_size = 2 bytes) global -> LDS (ISA ch.8 D#) ----
__device__ __forceinline__ void tdm_load_2d(unsigned lds_addr, const void* gaddr,
                                            unsigned tile0, unsigned tile1, unsigned td0,
                                            unsigned td1, unsigned stride0) {
  unsigned long long ga = (unsigned long long)(uintptr_t)gaddr;
  u32x4 g0;
  g0[0] = 1u;                                    // count=1, user descriptor
  g0[1] = lds_addr;                              // LDS byte address
  g0[2] = (unsigned)ga;                          // global_addr[31:0]
  g0[3] = (unsigned)(ga >> 32) | 0x80000000u;    // global_addr[56:32], type=2
  i32x8 g1;
  g1[0] = 0x10000;                               // data_size=1 -> 2 bytes
  g1[1] = (int)(td0 << 16);                      // tensor_dim0[15:0]
  g1[2] = (int)((td0 >> 16) | (td1 << 16));
  g1[3] = (int)((td1 >> 16) | (tile0 << 16));
  g1[4] = (int)tile1;                            // tile_dim1 (tile_dim2 = 0)
  g1[5] = (int)stride0;
  g1[6] = 0;
  g1[7] = 0;
  i32x4 z4 = {0, 0, 0, 0};
  i32x8 z8 = {0, 0, 0, 0, 0, 0, 0, 0};
  __builtin_amdgcn_tensor_load_to_lds(g0, g1, z4, z4, z8, 0);
}

// ---------------- Stage 1: BatchNorm statistics -> per-channel scale/shift ----------------
__global__ void k_bn(const float* __restrict__ x, const float* __restrict__ gamma,
                     const float* __restrict__ beta, float* __restrict__ sc,
                     float* __restrict__ sh) {
  __shared__ float r1[256], r2[256];
  const int c = blockIdx.x, tid = threadIdx.x;
  float s = 0.f, s2 = 0.f;
  for (int i = tid; i < 16384; i += 256) {
    float v = x[i * 64 + c];
    s += v;
    s2 += v * v;
  }
  r1[tid] = s;
  r2[tid] = s2;
  __syncthreads();
  for (int st = 128; st > 0; st >>= 1) {
    if (tid < st) {
      r1[tid] += r1[tid + st];
      r2[tid] += r2[tid + st];
    }
    __syncthreads();
  }
  if (tid == 0) {
    float mean = r1[0] * (1.f / 16384.f);
    float var  = r2[0] * (1.f / 16384.f) - mean * mean;
    float rstd = rsqrtf(var + 1e-5f);
    float scale = gamma[c] * rstd;
    sc[c] = scale;
    sh[c] = beta[c] - mean * scale;
  }
}

// ---------------- Stage 2: normalize + QKV projection (WMMA bf16) ----------------
// Q,K computed transposed (D = W^T x Xn^T): lane's 8 outputs contiguous along c_out
// -> uint4 global stores. Q pre-scaled by SM_SCALE. V normal: contiguous along key.
__global__ void k_qkv(const float* __restrict__ x, const float* __restrict__ w1,
                      const float* __restrict__ w2, const float* __restrict__ w3,
                      const float* __restrict__ b1, const float* __restrict__ b2,
                      const float* __restrict__ b3, const float* __restrict__ sc,
                      const float* __restrict__ sh, unsigned short* __restrict__ qws,
                      unsigned short* __restrict__ kws, unsigned short* __restrict__ vtws) {
  __shared__ unsigned short s_xn[128 * 64];
  __shared__ unsigned short s_wt[3][64 * 64];   // transposed: [c_out][c_in]
  const int tid = threadIdx.x;
  const int row0 = blockIdx.x * 128;

  for (int i = tid; i < 64 * 64; i += 256) {
    int t = (i & 63) * 64 + (i >> 6);           // transpose on LDS store
    s_wt[0][t] = f32_to_bf16(w1[i]);
    s_wt[1][t] = f32_to_bf16(w2[i]);
    s_wt[2][t] = f32_to_bf16(w3[i]);
  }
  for (int i = tid; i < 128 * 64; i += 256) {
    int c = i & 63;
    s_xn[i] = f32_to_bf16(x[row0 * 64 + i] * sc[c] + sh[c]);
  }
  __syncthreads();

  const int wave = tid >> 5, lane = tid & 31;
  const int half = lane >> 4, ln = lane & 15;
  const unsigned short* xn = s_xn + wave * 16 * 64;

  // ---- Q and K, transposed orientation: lane n = row, VGPR r = c_out ----
  v16bf xb0 = load_matBt(xn, 64, 0, lane);
  v16bf xb1 = load_matBt(xn, 64, 32, lane);
  const int row = row0 + wave * 16 + ln;
  for (int mat = 0; mat < 2; ++mat) {
    const float* bias = mat ? b2 : b1;
    unsigned short* dst = mat ? kws : qws;
    const float qs = mat ? 1.0f : SM_SCALE;     // fold softmax scale into Q
#pragma unroll
    for (int mt = 0; mt < 4; ++mt) {
      v8f acc = {};
      acc = WMMA_BF16(load_matA(s_wt[mat] + mt * 16 * 64, 64, 0, lane), xb0, acc);
      acc = WMMA_BF16(load_matA(s_wt[mat] + mt * 16 * 64, 64, 32, lane), xb1, acc);
      const float4* pb = (const float4*)(bias + mt * 16 + 8 * half);
      float4 bA = pb[0], bB = pb[1];
      uint4 w;
      w.x = pack_bf16x2((acc[0] + bA.x) * qs, (acc[1] + bA.y) * qs);
      w.y = pack_bf16x2((acc[2] + bA.z) * qs, (acc[3] + bA.w) * qs);
      w.z = pack_bf16x2((acc[4] + bB.x) * qs, (acc[5] + bB.y) * qs);
      w.w = pack_bf16x2((acc[6] + bB.z) * qs, (acc[7] + bB.w) * qs);
      *(uint4*)(dst + (size_t)row * 64 + mt * 16 + 8 * half) = w;
    }
  }

  // ---- V, normal orientation: lane n = chan, VGPR r = row -> contiguous keys ----
  v16bf xa0 = load_matA(xn, 64, 0, lane);
  v16bf xa1 = load_matA(xn, 64, 32, lane);
  const int bb_ = row0 >> 12;
  const int nbase = (row0 & 4095) + wave * 16 + 8 * half;
#pragma unroll
  for (int nt = 0; nt < 4; ++nt) {
    v8f acc = {};
    acc = WMMA_BF16(xa0, load_matBt(s_wt[2] + nt * 16 * 64, 64, 0, lane), acc);
    acc = WMMA_BF16(xa1, load_matBt(s_wt[2] + nt * 16 * 64, 64, 32, lane), acc);
    int c = nt * 16 + ln;
    float bb = b3[c];
    uint4 w;
    w.x = pack_bf16x2(acc[0] + bb, acc[1] + bb);
    w.y = pack_bf16x2(acc[2] + bb, acc[3] + bb);
    w.z = pack_bf16x2(acc[4] + bb, acc[5] + bb);
    w.w = pack_bf16x2(acc[6] + bb, acc[7] + bb);
    *(uint4*)(vtws + ((size_t)(bb_ * 64 + c)) * 4096 + nbase) = w;
  }
}

// ---------------- Stage 3: flash attention + fused output projection + residual ----------
// S^T = K Q^T (lane = query, keys in VGPRs). Max-free streaming softmax: scores are
// BN-normalized, |s|<<126 in log2 domain -> p=exp2(s) cannot overflow f32; divide by l
// at the end. Removes all per-iteration max/rescale work.
__global__ void k_attn(const unsigned short* __restrict__ qws,
                       const unsigned short* __restrict__ kws,
                       const unsigned short* __restrict__ vtws, const float* __restrict__ w4,
                       const float* __restrict__ b4, const float* __restrict__ x_in,
                       float* __restrict__ out) {
  __shared__ unsigned short s_k[2][64 * 64];    // [key][chan], double buffered
  __shared__ unsigned short s_vt[2][64 * 64];   // [chan][key], double buffered
  __shared__ unsigned short s_w4t[64 * 64];     // [c_out][c_in]
  __shared__ unsigned short s_p[8][16 * 64];    // per-wave P/O scratch [query][key|chan]

  const int tid = threadIdx.x, wave = tid >> 5, lane = tid & 31;
  const int half = lane >> 4, ln = lane & 15;
  const int b = blockIdx.x >> 5;
  const int qblk = blockIdx.x & 31;
  const int base = b * 4096;
  const int qrow0 = base + qblk * 128 + wave * 16;

  for (int i = tid; i < 64 * 64; i += 256)
    s_w4t[i] = f32_to_bf16(w4[(i & 63) * 64 + (i >> 6)]);

  // Q tile as B operand ([query][chan] row-major), loaded once (pre-scaled).
  v16bf qb0 = load_matBt(qws + (size_t)qrow0 * 64, 64, 0, lane);
  v16bf qb1 = load_matBt(qws + (size_t)qrow0 * 64, 64, 32, lane);

  v8f o[4] = {};
  float l = 0.f;                                // per-lane row-sum for query `ln`

  if (wave == 0) {                              // TDM prologue: stage block 0
    tdm_load_2d((unsigned)(uintptr_t)&s_k[0][0], kws + (size_t)base * 64,
                4096, 1, 262144u * 4u, 1, 4096);
    tdm_load_2d((unsigned)(uintptr_t)&s_vt[0][0], vtws + (size_t)b * 64 * 4096,
                64, 64, 4096, 64, 4096);
  }

  for (int kb = 0; kb < 64; ++kb) {
    const int cur = kb & 1;
    if (wave == 0) __builtin_amdgcn_s_wait_tensorcnt(0);
    __syncthreads();
    if (wave == 0 && kb + 1 < 64) {             // stage next block (other buffer)
      tdm_load_2d((unsigned)(uintptr_t)&s_k[cur ^ 1][0],
                  kws + (size_t)(base + (kb + 1) * 64) * 64, 4096, 1, 262144u * 4u, 1, 4096);
      tdm_load_2d((unsigned)(uintptr_t)&s_vt[cur ^ 1][0],
                  vtws + (size_t)b * 64 * 4096 + (kb + 1) * 64, 64, 64, 4096, 64, 4096);
    }

    // S^T = K (Q*scale)^T in log2 domain: lane = query ln; s[nt][r] = key nt*16+8hf+r
    v8f s[4];
#pragma unroll
    for (int nt = 0; nt < 4; ++nt) {
      v8f acc = {};
      acc = WMMA_BF16(load_matA(&s_k[cur][nt * 16 * 64], 64, 0, lane), qb0, acc);
      acc = WMMA_BF16(load_matA(&s_k[cur][nt * 16 * 64], 64, 32, lane), qb1, acc);
      s[nt] = acc;
    }

    float psum = 0.f;
#pragma unroll
    for (int nt = 0; nt < 4; ++nt)
#pragma unroll
      for (int r = 0; r < 8; ++r) {
        float p = exp2f(s[nt][r]);
        s[nt][r] = p;
        psum += p;
      }
    psum += __shfl_xor(psum, 16, 32);           // merge the two key-halves
    l += psum;

    // P -> wave-private LDS [query][key]: contiguous 16B runs -> b128 stores
    unsigned short* pw = s_p[wave];
#pragma unroll
    for (int nt = 0; nt < 4; ++nt) {
      uint4 w;
      w.x = pack_bf16x2(s[nt][0], s[nt][1]);
      w.y = pack_bf16x2(s[nt][2], s[nt][3]);
      w.z = pack_bf16x2(s[nt][4], s[nt][5]);
      w.w = pack_bf16x2(s[nt][6], s[nt][7]);
      *(uint4*)(pw + ln * 64 + nt * 16 + 8 * half) = w;
    }
    __threadfence_block();
    v16bf pa0 = load_matA(pw, 64, 0, lane);
    v16bf pa1 = load_matA(pw, 64, 32, lane);
#pragma unroll
    for (int nt = 0; nt < 4; ++nt) {            // O += P @ V (V^T stored [chan][key])
      o[nt] = WMMA_BF16(pa0, load_matBt(&s_vt[cur][nt * 16 * 64], 64, 0, lane), o[nt]);
      o[nt] = WMMA_BF16(pa1, load_matBt(&s_vt[cur][nt * 16 * 64], 64, 32, lane), o[nt]);
    }
  }

  // normalize: 1/l for query r+8*half lives in lane r+8*half
  float linv = 1.0f / l;
  float lv[8];
#pragma unroll
  for (int r = 0; r < 8; ++r) lv[r] = __shfl(linv, r + 8 * half, 32);
#pragma unroll
  for (int nt = 0; nt < 4; ++nt)
#pragma unroll
    for (int r = 0; r < 8; ++r) o[nt][r] *= lv[r];

  // fused epilogue: out = (O @ w4) + b4 + x_in
  unsigned short* pw = s_p[wave];
#pragma unroll
  for (int nt = 0; nt < 4; ++nt)
#pragma unroll
    for (int r = 0; r < 8; ++r)
      pw[(r + 8 * half) * 64 + nt * 16 + ln] = f32_to_bf16(o[nt][r]);
  __threadfence_block();
  v16bf oa0 = load_matA(pw, 64, 0, lane);
  v16bf oa1 = load_matA(pw, 64, 32, lane);
#pragma unroll
  for (int nt = 0; nt < 4; ++nt) {
    v8f d = {};
    d = WMMA_BF16(oa0, load_matBt(s_w4t + nt * 16 * 64, 64, 0, lane), d);
    d = WMMA_BF16(oa1, load_matBt(s_w4t + nt * 16 * 64, 64, 32, lane), d);
    float bb = b4[nt * 16 + ln];
#pragma unroll
    for (int r = 0; r < 8; ++r) {
      int idx = (qrow0 + r + 8 * half) * 64 + nt * 16 + ln;
      out[idx] = d[r] + bb + x_in[idx];
    }
  }
}

extern "C" void kernel_launch(void* const* d_in, const int* in_sizes, int n_in, void* d_out,
                              int out_size, void* d_ws, size_t ws_size, hipStream_t stream) {
  const float* x_in  = (const float*)d_in[0];
  const float* w1    = (const float*)d_in[1];
  const float* w2    = (const float*)d_in[2];
  const float* w3    = (const float*)d_in[3];
  const float* w4    = (const float*)d_in[4];
  const float* b1    = (const float*)d_in[5];
  const float* b2    = (const float*)d_in[6];
  const float* b3    = (const float*)d_in[7];
  const float* b4    = (const float*)d_in[8];
  const float* gamma = (const float*)d_in[9];
  const float* beta  = (const float*)d_in[10];
  float* out = (float*)d_out;

  char* ws = (char*)d_ws;
  float* sc = (float*)ws;
  float* sh = sc + 64;
  unsigned short* qws  = (unsigned short*)(ws + 1024);
  unsigned short* kws  = qws + (size_t)16384 * 64;
  unsigned short* vtws = kws + (size_t)16384 * 64;

  k_bn<<<64, 256, 0, stream>>>(x_in, gamma, beta, sc, sh);
  k_qkv<<<128, 256, 0, stream>>>(x_in, w1, w2, w3, b1, b2, b3, sc, sh, qws, kws, vtws);
  k_attn<<<128, 256, 0, stream>>>(qws, kws, vtws, w4, b4, x_in, out);
}
